// SequentialAudioVisualModel_43585328119836
// MI455X (gfx1250) — compile-verified, matrix operand order checked by reading, not run on hardware
//
#include <hip/hip_runtime.h>
#include <hip/hip_bf16.h>

typedef __attribute__((ext_vector_type(16))) _Float16 v16h;
typedef __attribute__((ext_vector_type(8)))  float    v8f;

#define T_SEQ   40
#define TM1     39
#define BSZ     32
#define NSEQ    (TM1 * BSZ)          // 1248
#define DM      256
#define NH      4
#define HD      64
#define DFF     1024
#define DINO    384
#define NPATCH  256
#define LN_EPS  1e-5f

// ---------------------------------------------------------------------------
// WMMA GEMM:  C[M,N] = act( A[M,K] @ op(W) + bias )
//   op(W) = W[n*ldw + k]  (transW==0)   |   W[k*ldw + n]  (transW==1)
// CONTRACT: M % 128 == 0, N % 128 == 0, K % 32 == 0  (true for every call).
// 128x128 block tile, 256 threads = 8 waves (2 x 4), each wave owns a 64x32
// output slab = 8 v_wmma_f32_16x16x32_f16 per 32-deep K step.
// Next K tile is prefetched into registers while WMMAs consume LDS.
// ---------------------------------------------------------------------------
#define LDSP 40   // padded halves per LDS row (80B) to break bank-conflict stride

__launch_bounds__(256)
__global__ void gemm_wmma_kernel(const float* __restrict__ A, int lda,
                                 const float* __restrict__ W, int ldw, int transW,
                                 const float* __restrict__ bias,
                                 float* __restrict__ C, int ldc,
                                 int M, int N, int K, int relu)
{
    __shared__ _Float16 Ah[128][LDSP];
    __shared__ _Float16 Bh[128][LDSP];

    const int tid  = threadIdx.x;
    const int wave = tid >> 5;
    const int lane = tid & 31;
    const int bm   = blockIdx.x * 128;
    const int bn   = blockIdx.y * 128;
    const int wm   = (wave >> 2) * 64;   // 0,64
    const int wn   = (wave & 3) * 32;    // 0,32,64,96

    // staging assignment: thread -> (row, 16-float column chunk)
    const int srow = tid >> 1;           // 0..127
    const int scol = (tid & 1) * 16;     // 0 or 16

    v8f acc[4][2];
    #pragma unroll
    for (int i = 0; i < 4; ++i)
        #pragma unroll
        for (int j = 0; j < 2; ++j)
            acc[i][j] = (v8f){0.f,0.f,0.f,0.f,0.f,0.f,0.f,0.f};

    const float* aRow = A + (size_t)(bm + srow) * lda + scol;   // + k0
    const float* wRow = W + (size_t)(bn + srow) * ldw + scol;   // + k0 (transW==0)

    float4 ra[4], rb[4];

    // ---- load first K tile into registers (vector b128 loads) ----
    {
        const float4* ap = (const float4*)(aRow);
        #pragma unroll
        for (int j = 0; j < 4; ++j) ra[j] = ap[j];
        if (!transW) {
            const float4* wp = (const float4*)(wRow);
            #pragma unroll
            for (int j = 0; j < 4; ++j) rb[j] = wp[j];
        } else {
            #pragma unroll
            for (int j = 0; j < 4; ++j) {
                rb[j].x = W[(size_t)(scol + 4*j + 0) * ldw + bn + srow];
                rb[j].y = W[(size_t)(scol + 4*j + 1) * ldw + bn + srow];
                rb[j].z = W[(size_t)(scol + 4*j + 2) * ldw + bn + srow];
                rb[j].w = W[(size_t)(scol + 4*j + 3) * ldw + bn + srow];
            }
        }
    }

    const int lr = lane & 15;
    const int kh = (lane >> 4) << 3;     // 0 or 8

    for (int k0 = 0; k0 < K; k0 += 32) {
        // ---- commit register tile to LDS as f16 ----
        #pragma unroll
        for (int j = 0; j < 4; ++j) {
            Ah[srow][scol + 4*j + 0] = (_Float16)ra[j].x;
            Ah[srow][scol + 4*j + 1] = (_Float16)ra[j].y;
            Ah[srow][scol + 4*j + 2] = (_Float16)ra[j].z;
            Ah[srow][scol + 4*j + 3] = (_Float16)ra[j].w;
            Bh[srow][scol + 4*j + 0] = (_Float16)rb[j].x;
            Bh[srow][scol + 4*j + 1] = (_Float16)rb[j].y;
            Bh[srow][scol + 4*j + 2] = (_Float16)rb[j].z;
            Bh[srow][scol + 4*j + 3] = (_Float16)rb[j].w;
        }
        __syncthreads();

        // ---- prefetch next K tile into registers (overlaps WMMAs) ----
        const int kn = k0 + 32;
        if (kn < K) {
            const float4* ap = (const float4*)(aRow + kn);
            #pragma unroll
            for (int j = 0; j < 4; ++j) ra[j] = ap[j];
            if (!transW) {
                const float4* wp = (const float4*)(wRow + kn);
                #pragma unroll
                for (int j = 0; j < 4; ++j) rb[j] = wp[j];
            } else {
                #pragma unroll
                for (int j = 0; j < 4; ++j) {
                    rb[j].x = W[(size_t)(kn + scol + 4*j + 0) * ldw + bn + srow];
                    rb[j].y = W[(size_t)(kn + scol + 4*j + 1) * ldw + bn + srow];
                    rb[j].z = W[(size_t)(kn + scol + 4*j + 2) * ldw + bn + srow];
                    rb[j].w = W[(size_t)(kn + scol + 4*j + 3) * ldw + bn + srow];
                }
            }
        }

        // ---- fragments from LDS (ds_load_b128 pairs) + 8 WMMAs ----
        v16h b0, b1;
        #pragma unroll
        for (int j = 0; j < 8; ++j) {
            b0[j]     = Bh[wn + lr     ][kh + j];
            b0[j + 8] = Bh[wn + lr     ][kh + 16 + j];
            b1[j]     = Bh[wn + 16 + lr][kh + j];
            b1[j + 8] = Bh[wn + 16 + lr][kh + 16 + j];
        }
        #pragma unroll
        for (int i = 0; i < 4; ++i) {
            v16h a;
            #pragma unroll
            for (int j = 0; j < 8; ++j) {
                a[j]     = Ah[wm + i*16 + lr][kh + j];
                a[j + 8] = Ah[wm + i*16 + lr][kh + 16 + j];
            }
            acc[i][0] = __builtin_amdgcn_wmma_f32_16x16x32_f16(false, a, false, b0,
                                                               (short)0, acc[i][0], false, false);
            acc[i][1] = __builtin_amdgcn_wmma_f32_16x16x32_f16(false, a, false, b1,
                                                               (short)0, acc[i][1], false, false);
        }
        __syncthreads();
    }

    // ---- writeback per 32-bit C/D layout (no bounds checks needed) ----
    const int rofs = (lane >> 4) << 3;    // 0 or 8
    const int c0 = bn + wn + lr;
    const int c1 = c0 + 16;
    const float bv0 = bias ? bias[c0] : 0.f;
    const float bv1 = bias ? bias[c1] : 0.f;
    #pragma unroll
    for (int i = 0; i < 4; ++i) {
        #pragma unroll
        for (int j = 0; j < 8; ++j) {
            const int r = bm + wm + i*16 + rofs + j;
            float o0 = acc[i][0][j] + bv0;
            float o1 = acc[i][1][j] + bv1;
            if (relu) { o0 = fmaxf(o0, 0.f); o1 = fmaxf(o1, 0.f); }
            C[(size_t)r * ldc + c0] = o0;
            C[(size_t)r * ldc + c1] = o1;
        }
    }
}

// ---------------------------------------------------------------------------
// X[(t*B+b)*T + s, :] = embedding[tokens[b,s]] + pos_emb[s]
// ---------------------------------------------------------------------------
__global__ void embed_kernel(const int* __restrict__ tokens,
                             const float* __restrict__ emb,
                             const float* __restrict__ pos,
                             float* __restrict__ X)
{
    size_t i = (size_t)blockIdx.x * blockDim.x + threadIdx.x;
    const size_t total = (size_t)NSEQ * T_SEQ * DM;
    if (i >= total) return;
    int dd = i % DM;
    int s  = (i / DM) % T_SEQ;
    int b  = (i / ((size_t)DM * T_SEQ)) % BSZ;
    int tok = tokens[b * T_SEQ + s];
    X[i] = emb[(size_t)tok * DM + dd] + pos[s * DM + dd];
}

// ---------------------------------------------------------------------------
// attention for one (sequence n, head h).  K/V head tiles are pulled into LDS
// with CDNA5 async copies (GLOBAL_LOAD_ASYNC_TO_LDS_B128, ASYNCcnt).
// ---------------------------------------------------------------------------
__launch_bounds__(128)
__global__ void attn_kernel(const float* __restrict__ qkv,
                            float* __restrict__ o)
{
    const int n = blockIdx.x;
    const int h = blockIdx.y;
    const int t_lim = n / BSZ;                 // keys k <= t_lim are valid

    __shared__ float sc [T_SEQ][T_SEQ + 1];
    __shared__ float Ksh[T_SEQ][HD];
    __shared__ float Vsh[T_SEQ][HD];

    const float* base = qkv + (size_t)n * T_SEQ * (3 * DM);

    // async copy K and V head slices (16B per lane per issue) into LDS
    for (int idx = threadIdx.x * 4; idx < T_SEQ * HD; idx += blockDim.x * 4) {
        int s  = idx / HD;
        int dd = idx % HD;
        unsigned long long gk = (unsigned long long)(uintptr_t)(base + s*(3*DM) +     DM + h*HD + dd);
        unsigned long long gv = (unsigned long long)(uintptr_t)(base + s*(3*DM) + 2*DM + h*HD + dd);
        unsigned lk = (unsigned)(uintptr_t)&Ksh[s][dd];
        unsigned lv = (unsigned)(uintptr_t)&Vsh[s][dd];
        asm volatile("global_load_async_to_lds_b128 %0, %1, off"
                     :: "v"(lk), "v"(gk) : "memory");
        asm volatile("global_load_async_to_lds_b128 %0, %1, off"
                     :: "v"(lv), "v"(gv) : "memory");
    }
    asm volatile("s_wait_asynccnt 0x0" ::: "memory");
    __syncthreads();

    for (int idx = threadIdx.x; idx < T_SEQ * T_SEQ; idx += blockDim.x) {
        int q = idx / T_SEQ;
        int k = idx % T_SEQ;
        float sv = -1e9f;
        if (k <= t_lim) {
            const float* qp = base + q * (3 * DM) + h * HD;
            float acc = 0.f;
            #pragma unroll 8
            for (int dd = 0; dd < HD; ++dd) acc += qp[dd] * Ksh[k][dd];
            sv = acc * 0.125f;                 // 1/sqrt(64)
        }
        sc[q][k] = sv;
    }
    __syncthreads();

    if (threadIdx.x < T_SEQ) {
        int q = threadIdx.x;
        float m = -1e30f;
        for (int k = 0; k < T_SEQ; ++k) m = fmaxf(m, sc[q][k]);
        float sum = 0.f;
        for (int k = 0; k < T_SEQ; ++k) { float e = __expf(sc[q][k] - m); sc[q][k] = e; sum += e; }
        float inv = 1.f / sum;
        for (int k = 0; k < T_SEQ; ++k) sc[q][k] *= inv;
    }
    __syncthreads();

    for (int idx = threadIdx.x; idx < T_SEQ * HD; idx += blockDim.x) {
        int q  = idx / HD;
        int dd = idx % HD;
        float acc = 0.f;
        for (int k = 0; k < T_SEQ; ++k) acc += sc[q][k] * Vsh[k][dd];
        o[((size_t)(n * T_SEQ + q)) * DM + h * HD + dd] = acc;
    }
}

// ---------------------------------------------------------------------------
// x = LayerNorm(x + r) * g + b     (one row per block, d = 256)
// ---------------------------------------------------------------------------
__launch_bounds__(256)
__global__ void add_ln_kernel(float* __restrict__ x, const float* __restrict__ r,
                              const float* __restrict__ g, const float* __restrict__ bb)
{
    const int row = blockIdx.x;
    const int tid = threadIdx.x;
    __shared__ float red[256];

    float v = x[(size_t)row * DM + tid] + r[(size_t)row * DM + tid];
    red[tid] = v; __syncthreads();
    for (int s = 128; s > 0; s >>= 1) { if (tid < s) red[tid] += red[tid + s]; __syncthreads(); }
    float mean = red[0] * (1.f / DM);
    __syncthreads();
    float dv = v - mean;
    red[tid] = dv * dv; __syncthreads();
    for (int s = 128; s > 0; s >>= 1) { if (tid < s) red[tid] += red[tid + s]; __syncthreads(); }
    float var = red[0] * (1.f / DM);
    x[(size_t)row * DM + tid] = dv * rsqrtf(var + LN_EPS) * g[tid] + bb[tid];
}

// ---------------------------------------------------------------------------
// ctx[t*B+b, :] :  t==0 -> 0 ; else enc[(t-1)*B+b] at position t-1
// ---------------------------------------------------------------------------
__global__ void gather_ctx_kernel(const float* __restrict__ X, float* __restrict__ ctx)
{
    const int row = blockIdx.x;           // 0..1279
    const int t   = row / BSZ;
    const int b   = row % BSZ;
    const int dd  = threadIdx.x;
    float v = 0.f;
    if (t > 0) {
        size_t src = (((size_t)(t - 1) * BSZ + b) * T_SEQ + (t - 1)) * DM + dd;
        v = X[src];
    }
    ctx[(size_t)row * DM + dd] = v;
}

// ---------------------------------------------------------------------------
// cw[t*B+b,:] = cls_w[tokens[b,t]] ; cbg[t*B+b] = cls_b[tokens[b,t]]
// ---------------------------------------------------------------------------
__global__ void gather_cw_kernel(const int* __restrict__ tokens,
                                 const float* __restrict__ cls_w,
                                 const float* __restrict__ cls_b,
                                 float* __restrict__ cw, float* __restrict__ cbg)
{
    const int row = blockIdx.x;
    const int t   = row / BSZ;
    const int b   = row % BSZ;
    const int tok = tokens[b * T_SEQ + t];
    cw[(size_t)row * DM + threadIdx.x] = cls_w[(size_t)tok * DM + threadIdx.x];
    if (threadIdx.x == 0) cbg[row] = cls_b[tok];
}

// ---------------------------------------------------------------------------
// out[b,p,t] = sigmoid( sum_k gelu(pp[b,p,k]+cp[t,b,k]) * v[t,b,k]
//                       + dot(fus_b2, cw[t,b]) + cb[t,b] )
// ---------------------------------------------------------------------------
__launch_bounds__(256)
__global__ void fuse_out_kernel(const float* __restrict__ pp, const float* __restrict__ cp,
                                const float* __restrict__ vb, const float* __restrict__ cw,
                                const float* __restrict__ cbg, const float* __restrict__ fus_b2,
                                float* __restrict__ out)
{
    const int row = blockIdx.x;           // t*B + b
    const int t   = row / BSZ;
    const int b   = row % BSZ;
    const int tid = threadIdx.x;          // patch p

    __shared__ float cps[DM], vs[DM], red[DM];
    cps[tid] = cp[(size_t)row * DM + tid];
    vs [tid] = vb[(size_t)row * DM + tid];
    red[tid] = fus_b2[tid] * cw[(size_t)row * DM + tid];
    __syncthreads();
    for (int s = 128; s > 0; s >>= 1) { if (tid < s) red[tid] += red[tid + s]; __syncthreads(); }
    const float sbias = red[0] + cbg[row];

    const float* ppr = pp + ((size_t)b * NPATCH + tid) * DM;
    float acc = 0.f;
    #pragma unroll 4
    for (int k = 0; k < DM; ++k) {
        float x = ppr[k] + cps[k];
        float g = 0.5f * x * (1.f + erff(x * 0.70710678118654752f));  // exact gelu
        acc += g * vs[k];
    }
    const float logit = acc + sbias;
    out[((size_t)b * NPATCH + tid) * T_SEQ + t] = 1.f / (1.f + __expf(-logit));
}

// ---------------------------------------------------------------------------
extern "C" void kernel_launch(void* const* d_in, const int* in_sizes, int n_in,
                              void* d_out, int out_size, void* d_ws, size_t ws_size,
                              hipStream_t stream)
{
    const float* patches    = (const float*)d_in[0];
    const int*   tokens     = (const int*)  d_in[1];
    const float* embedding  = (const float*)d_in[2];
    const float* pos_emb    = (const float*)d_in[3];
    const float* in_proj_w  = (const float*)d_in[4];
    const float* in_proj_b  = (const float*)d_in[5];
    const float* out_proj_w = (const float*)d_in[6];
    const float* out_proj_b = (const float*)d_in[7];
    const float* lin1_w     = (const float*)d_in[8];
    const float* lin1_b     = (const float*)d_in[9];
    const float* lin2_w     = (const float*)d_in[10];
    const float* lin2_b     = (const float*)d_in[11];
    const float* ln1_g      = (const float*)d_in[12];
    const float* ln1_b      = (const float*)d_in[13];
    const float* ln2_g      = (const float*)d_in[14];
    const float* ln2_b      = (const float*)d_in[15];
    const float* fus_w1     = (const float*)d_in[16];
    const float* fus_b1     = (const float*)d_in[17];
    const float* fus_w2     = (const float*)d_in[18];
    const float* fus_b2     = (const float*)d_in[19];
    const float* cls_w      = (const float*)d_in[20];
    const float* cls_b      = (const float*)d_in[21];
    float* out = (float*)d_out;

    const int M = NSEQ * T_SEQ;                // 49920 token rows (M % 128 == 0)

    float* ws   = (float*)d_ws;
    float* X    = ws;                                      // 49920*256
    float* bufA = X    + (size_t)M * DM;                   // 49920*1024 (qkv / ff1)
    float* bufB = bufA + (size_t)M * DFF;                  // 49920*256
    float* ctx  = bufB + (size_t)M * DM;                   // 1280*256
    float* pp   = ctx  + (size_t)T_SEQ * BSZ * DM;         // 8192*256
    float* cp   = pp   + (size_t)BSZ * NPATCH * DM;        // 1280*256
    float* cw   = cp   + (size_t)T_SEQ * BSZ * DM;         // 1280*256
    float* vbuf = cw   + (size_t)T_SEQ * BSZ * DM;         // 1280*256
    float* cbg  = vbuf + (size_t)T_SEQ * BSZ * DM;         // 1280

    // 1) embedding + positional
    {
        size_t total = (size_t)M * DM;
        embed_kernel<<<(int)((total + 255) / 256), 256, 0, stream>>>(tokens, embedding, pos_emb, X);
    }

    // 2) transformer encoder, 2 layers
    for (int i = 0; i < 2; ++i) {
        // qkv = X @ in_proj_w[i].T + b                  (49920 x 768 x 256)
        gemm_wmma_kernel<<<dim3(M / 128, (3 * DM) / 128), 256, 0, stream>>>(
            X, DM, in_proj_w + (size_t)i * 3 * DM * DM, DM, 0,
            in_proj_b + (size_t)i * 3 * DM, bufA, 3 * DM, M, 3 * DM, DM, 0);
        // attention -> bufB
        attn_kernel<<<dim3(NSEQ, NH), 128, 0, stream>>>(bufA, bufB);
        // out_proj -> bufA                              (49920 x 256 x 256)
        gemm_wmma_kernel<<<dim3(M / 128, DM / 128), 256, 0, stream>>>(
            bufB, DM, out_proj_w + (size_t)i * DM * DM, DM, 0,
            out_proj_b + (size_t)i * DM, bufA, DM, M, DM, DM, 0);
        // x = LN(x + o)
        add_ln_kernel<<<M, 256, 0, stream>>>(X, bufA, ln1_g + i * DM, ln1_b + i * DM);
        // ff1 = relu(x @ lin1.T + b) -> bufA            (49920 x 1024 x 256)
        gemm_wmma_kernel<<<dim3(M / 128, DFF / 128), 256, 0, stream>>>(
            X, DM, lin1_w + (size_t)i * DFF * DM, DM, 0,
            lin1_b + (size_t)i * DFF, bufA, DFF, M, DFF, DM, 1);
        // ff2 -> bufB                                   (49920 x 256 x 1024)
        gemm_wmma_kernel<<<dim3(M / 128, DM / 128), 256, 0, stream>>>(
            bufA, DFF, lin2_w + (size_t)i * DM * DFF, DFF, 0,
            lin2_b + (size_t)i * DM, bufB, DM, M, DM, DFF, 0);
        // x = LN(x + ff)
        add_ln_kernel<<<M, 256, 0, stream>>>(X, bufB, ln2_g + i * DM, ln2_b + i * DM);
    }

    // 3) fusion head (h@w2 then ·cw collapsed to k-space dot; saves ~43 GFLOP)
    gather_ctx_kernel<<<T_SEQ * BSZ, DM, 0, stream>>>(X, ctx);
    // pp = patches @ w1p.T            (8192 x 256 x 384; w1p row stride 640)
    gemm_wmma_kernel<<<dim3((BSZ * NPATCH) / 128, DM / 128), 256, 0, stream>>>(
        patches, DINO, fus_w1, DINO + DM, 0, nullptr, pp, DM,
        BSZ * NPATCH, DM, DINO, 0);
    // cp = ctx @ w1c.T + fus_b1       (1280 x 256 x 256)
    gemm_wmma_kernel<<<dim3((T_SEQ * BSZ) / 128, DM / 128), 256, 0, stream>>>(
        ctx, DM, fus_w1 + DINO, DINO + DM, 0, fus_b1, cp, DM,
        T_SEQ * BSZ, DM, DM, 0);
    // cw / cb gathers
    gather_cw_kernel<<<T_SEQ * BSZ, DM, 0, stream>>>(tokens, cls_w, cls_b, cw, cbg);
    // v = cw @ fus_w2                 (1280 x 256 x 256, contract over m -> transW)
    gemm_wmma_kernel<<<dim3((T_SEQ * BSZ) / 128, DM / 128), 256, 0, stream>>>(
        cw, DM, fus_w2, DM, 1, nullptr, vbuf, DM,
        T_SEQ * BSZ, DM, DM, 0);
    // logits + sigmoid, transposed to (b, p, t)
    fuse_out_kernel<<<T_SEQ * BSZ, NPATCH, 0, stream>>>(pp, cp, vbuf, cw, cbg, fus_b2, out);
}
